// CRFDecoder_66915590472343
// MI455X (gfx1250) — compile-verified, matrix operand order checked by reading, not run on hardware
//
#include <hip/hip_runtime.h>
#include <math.h>

// CRF loss for MI455X (gfx1250, wave32).
// Forward scan: per workgroup = 16 batch rows, 16 waves; each step is
// P(16x256) = A(16x256,f16) x (exp(T)-1)(256x256,f16) via 16x8 chained
// v_wmma_f32_16x16x32_f16, then s' = m + log(rowsum + P) + em[t].
// exp(T) = 1 + D split keeps fp32 accuracy while using the f16 WMMA path.

typedef __attribute__((ext_vector_type(16))) _Float16 v16h;
typedef __attribute__((ext_vector_type(8)))  float    v8f;

#define SLN 512
#define BSZ 128
#define TAG 256
#define BT  16   // batch rows per workgroup
#define NW  16   // waves per workgroup (one 16-wide tag tile each)

// Half-index -> K mapping inside a 16-bit 16x32 (A) / 32x16 (B) WMMA fragment
// (CDNA5 ISA 7.12.2): VGPR0..3 hold K = hi*8 + 0..7,
// VGPR4..7 hold K = 16 + hi*8 + 0..7, where hi = lane/16.
__device__ __forceinline__ int kidx(int e, int hi) {
    return (e < 8) ? (hi * 8 + e) : (16 + hi * 8 + (e - 8));
}

__global__ __launch_bounds__(NW * 32)
void crf_forward(const float* __restrict__ em, const int* __restrict__ length,
                 const float* __restrict__ T,  const float* __restrict__ startT,
                 const float* __restrict__ endT, float* __restrict__ logZ)
{
    __shared__ float sCur[BT * TAG];                  // authoritative state (fp32)
    __shared__ float sNew[BT * TAG];                  // candidate state this step
    __shared__ __align__(32) _Float16 aF[8 * 32 * 16]; // A fragments, WMMA layout
    __shared__ float mB[BT];                          // per-batch running max
    __shared__ float rsB[BT];                         // per-batch sum exp(s-m)

    const int tid  = threadIdx.x;
    const int w    = tid >> 5;        // wave id: tag tile (phase 1) / batch (phase 2)
    const int lane = tid & 31;
    const int hi   = lane >> 4;
    const int nn   = lane & 15;
    const int b0   = blockIdx.x * BT;
    const int j0   = w * 16;

    // Loop-invariant B fragments: B[k][n] = exp(T[k][j0+n]) - 1  (|.| <= ~1e-2)
    v16h bF[8];
    for (int kb = 0; kb < 8; ++kb) {
        v16h bf;
        for (int e = 0; e < 16; ++e) {
            int k = kb * 32 + kidx(e, hi);
            bf[e] = (_Float16)(expf(T[k * TAG + j0 + nn]) - 1.0f);
        }
        bF[kb] = bf;
    }
    const int lenw = length[b0 + w];  // batch this wave renormalizes in phase 2

    // t = 0: s = start + em[0];  build m, rowsum, A fragments (wave w <-> batch w)
    {
        float sv[8];
        float mx = -1e30f;
        for (int q = 0; q < 8; ++q) {
            int j = lane * 8 + q;
            float v = startT[j] + em[(size_t)(b0 + w) * TAG + j];
            sCur[w * TAG + j] = v;
            sv[q] = v;
            mx = fmaxf(mx, v);
        }
        for (int off = 16; off > 0; off >>= 1) mx = fmaxf(mx, __shfl_xor(mx, off));
        float rs = 0.f;
        for (int q = 0; q < 8; ++q) rs += expf(sv[q] - mx);
        for (int off = 16; off > 0; off >>= 1) rs += __shfl_xor(rs, off);
        if (lane == 0) { mB[w] = mx; rsB[w] = rs; }
        for (int q = 0; q < 8; ++q) {
            int j  = lane * 8 + q;
            int kb = j >> 5, k = j & 31;
            int h2 = (k >> 3) & 1;
            int e  = (k & 7) + ((k & 16) >> 1);
            aF[((kb * 32 + h2 * 16 + w) << 4) + e] = (_Float16)expf(sv[q] - mx);
        }
    }
    __syncthreads();

    for (int t = 1; t < SLN; ++t) {
        // ---- phase 1: WMMA correction  P = A x (exp(T)-1), all lanes active ----
        v8f c = {};
        for (int kb = 0; kb < 8; ++kb) {
            v16h a = *(const v16h*)(&aF[(kb * 32 + lane) << 4]);
            c = __builtin_amdgcn_wmma_f32_16x16x32_f16(
                    false, a, false, bF[kb], (short)0, c, false, false);
        }
        const int j = j0 + nn;
        if (t + 1 < SLN)  // gfx1250 prefetch of next timestep's emission slice
            __builtin_prefetch(&em[((size_t)(t + 1) * BSZ + b0 + hi * 8) * TAG + j], 0, 0);
        for (int r = 0; r < 8; ++r) {
            int m = hi * 8 + r;       // C-tile row -> batch row (VGPR layout)
            float val = mB[m] + logf(rsB[m] + c[r])
                      + em[((size_t)t * BSZ + b0 + m) * TAG + j];
            sNew[m * TAG + j] = val;
        }
        __syncthreads();

        // ---- phase 2: commit + renormalize (wave w owns batch w; pad => skip) ----
        if (t < lenw) {
            float sv[8];
            float mx = -1e30f;
            for (int q = 0; q < 8; ++q) {
                int jj = lane * 8 + q;
                float v = sNew[w * TAG + jj];
                sCur[w * TAG + jj] = v;
                sv[q] = v;
                mx = fmaxf(mx, v);
            }
            for (int off = 16; off > 0; off >>= 1) mx = fmaxf(mx, __shfl_xor(mx, off));
            float rs = 0.f;
            for (int q = 0; q < 8; ++q) rs += expf(sv[q] - mx);
            for (int off = 16; off > 0; off >>= 1) rs += __shfl_xor(rs, off);
            if (lane == 0) { mB[w] = mx; rsB[w] = rs; }
            for (int q = 0; q < 8; ++q) {
                int jj = lane * 8 + q;
                int kb = jj >> 5, k = jj & 31;
                int h2 = (k >> 3) & 1;
                int e  = (k & 7) + ((k & 16) >> 1);
                aF[((kb * 32 + h2 * 16 + w) << 4) + e] = (_Float16)expf(sv[q] - mx);
            }
        }
        __syncthreads();
    }

    // logZ_b = logsumexp_j(sCur + end)
    {
        float sv[8];
        float mx = -1e30f;
        for (int q = 0; q < 8; ++q) {
            int j = lane * 8 + q;
            float v = sCur[w * TAG + j] + endT[j];
            sv[q] = v;
            mx = fmaxf(mx, v);
        }
        for (int off = 16; off > 0; off >>= 1) mx = fmaxf(mx, __shfl_xor(mx, off));
        float rs = 0.f;
        for (int q = 0; q < 8; ++q) rs += expf(sv[q] - mx);
        for (int off = 16; off > 0; off >>= 1) rs += __shfl_xor(rs, off);
        if (lane == 0) logZ[b0 + w] = mx + logf(rs);
    }
}

__global__ void crf_score(const float* __restrict__ em, const int* __restrict__ length,
                          const int* __restrict__ target, const float* __restrict__ T,
                          const float* __restrict__ startT, const float* __restrict__ endT,
                          float* __restrict__ sc)
{
    int b = threadIdx.x;
    int len = length[b];                       // padding is prefix-contiguous: pad = t >= len
    int prev = target[b];                      // target[0, b]
    float acc = startT[prev] + em[(size_t)b * TAG + prev];
    for (int t = 1; t < SLN; ++t) {
        if (t >= len) break;
        int tg = target[t * BSZ + b];
        acc += em[((size_t)t * BSZ + b) * TAG + tg] + T[prev * TAG + tg];
        prev = tg;
    }
    acc += endT[prev];                         // prev == target[len-1, b]
    sc[b] = acc;
}

__global__ void crf_reduce(const float* __restrict__ lz, const float* __restrict__ sc,
                           float* __restrict__ out)
{
    __shared__ float buf[BSZ];
    int i = threadIdx.x;
    buf[i] = lz[i] - sc[i];
    __syncthreads();
    for (int s = BSZ / 2; s > 0; s >>= 1) {
        if (i < s) buf[i] += buf[i + s];
        __syncthreads();
    }
    if (i == 0) out[0] = buf[0];
}

extern "C" void kernel_launch(void* const* d_in, const int* in_sizes, int n_in,
                              void* d_out, int out_size, void* d_ws, size_t ws_size,
                              hipStream_t stream)
{
    const float* em     = (const float*)d_in[0];   // (512,128,256) f32
    const int*   length = (const int*)  d_in[1];   // (128,) i32
    // d_in[2] = padding_mask (unused: derived from length)
    const int*   target = (const int*)  d_in[3];   // (512,128) i32
    const float* T      = (const float*)d_in[4];   // (256,256) f32
    const float* startT = (const float*)d_in[5];   // (256,) f32
    const float* endT   = (const float*)d_in[6];   // (256,) f32

    float* lz = (float*)d_ws;        // 128 floats
    float* sc = lz + BSZ;            // 128 floats

    crf_forward<<<BSZ / BT, NW * 32, 0, stream>>>(em, length, T, startT, endT, lz);
    crf_score  <<<1, BSZ, 0, stream>>>(em, length, target, T, startT, endT, sc);
    crf_reduce <<<1, BSZ, 0, stream>>>(lz, sc, (float*)d_out);
}
